// TimeAwareTransformerPredictor_90022514524492
// MI455X (gfx1250) — compile-verified, matrix operand order checked by reading, not run on hardware
//
#include <hip/hip_runtime.h>
#include <hip/hip_bf16.h>

// ---------------------------------------------------------------------------
// Model dims (compile-time constants from the reference)
// ---------------------------------------------------------------------------
#define BB   16
#define LL   1024
#define IND  32
#define DD   256
#define HH   8
#define DHD  32
#define NLL  2
#define DFFD 1024
#define BLROWS (BB * LL)   // 16384

typedef __attribute__((ext_vector_type(16))) _Float16 v16h;
typedef __attribute__((ext_vector_type(8)))  _Float16 v8h;
typedef __attribute__((ext_vector_type(8)))  float    v8f;
typedef __attribute__((ext_vector_type(4)))  int      v4i;

union V16H { v16h v; v8h h[2]; };

// ---------------------------------------------------------------------------
// CDNA5 async global->LDS copy (device pass only, builtin-gated)
// Prototype (from hipcc diagnostic): (v4i addrspace(1)*, v4i addrspace(3)*, imm, imm)
// ---------------------------------------------------------------------------
#if defined(__AMDGCN__) && __has_builtin(__builtin_amdgcn_global_load_async_to_lds_b128)
#define USE_ASYNC_LDS 1
#else
#define USE_ASYNC_LDS 0
#endif

#if USE_ASYNC_LDS
typedef __attribute__((address_space(1))) v4i* gptr4;
typedef __attribute__((address_space(3))) v4i* lptr4;
__device__ inline void cp_async16(void* l, const void* g) {
    __builtin_amdgcn_global_load_async_to_lds_b128(
        (gptr4)g, (lptr4)l, /*offset=*/0, /*cpol=*/0);
}
__device__ inline void wait_async0() {
#if __has_builtin(__builtin_amdgcn_s_wait_asynccnt)
    __builtin_amdgcn_s_wait_asynccnt(0);
#else
    asm volatile("s_wait_asynccnt 0x0" ::: "memory");
#endif
}
#endif

// D = A(16x32 f16) * B(32x16 f16) + C(16x16 f32)
__device__ inline v8f wmma_f16(v16h a, v16h b, v8f c) {
    return __builtin_amdgcn_wmma_f32_16x16x32_f16(
        /*neg_a=*/false, a, /*neg_b=*/false, b,
        /*c_mod=*/(short)0, c, /*reuse_a=*/false, /*reuse_b=*/false);
}

// Load one 16x32 f16 operand tile (A-layout; also valid for B when the
// source matrix stores each WMMA-column as a contiguous 32-half row).
// Per ISA 7.12.2: lane = idx%16, group g = lane/16; halves 0..7 hold
// k = 8g..8g+7, halves 8..15 hold k = 16+8g..23+8g.
__device__ inline v16h ld_tile(const _Float16* base, int stride, int lane) {
    int r = lane & 15;
    int g = lane >> 4;
    const _Float16* p = base + (size_t)r * stride + g * 8;
    V16H u;
    u.h[0] = *(const v8h*)(p);
    u.h[1] = *(const v8h*)(p + 16);
    return u.v;
}

// ---------------------------------------------------------------------------
// t[b,l] = cumsum over l of features[b,l,IN-2]
// ---------------------------------------------------------------------------
__global__ void k_cumsum(const float* __restrict__ feat, float* __restrict__ t) {
    __shared__ float s[LL];
    int b = blockIdx.x, i = threadIdx.x;
    s[i] = feat[((size_t)b * LL + i) * IND + (IND - 2)];
    __syncthreads();
    for (int off = 1; off < LL; off <<= 1) {
        float v = (i >= off) ? s[i - off] : 0.f;
        __syncthreads();
        s[i] += v;
        __syncthreads();
    }
    t[(size_t)b * LL + i] = s[i];
}

// ---------------------------------------------------------------------------
// x = features @ in_w.T + in_b + pe  (f32 + f16 copies)
// ---------------------------------------------------------------------------
__global__ void k_input(const float* __restrict__ f, const float* __restrict__ in_w,
                        const float* __restrict__ in_b, const float* __restrict__ pe,
                        float* __restrict__ x, _Float16* __restrict__ xh) {
    int idx = blockIdx.x * 256 + threadIdx.x;           // over B*L*D
    int d  = idx & (DD - 1);
    int bl = idx >> 8;
    int l  = bl & (LL - 1);
    const float* fr = f + (size_t)bl * IND;
    const float* wr = in_w + (size_t)d * IND;
    float s = in_b[d] + pe[(size_t)l * DD + d];
    #pragma unroll
    for (int i = 0; i < IND; ++i) s += fr[i] * wr[i];
    x[idx] = s;
    xh[idx] = (_Float16)s;
}

__global__ void k_f32toh(const float* __restrict__ a, _Float16* __restrict__ o, int n) {
    int i = blockIdx.x * 256 + threadIdx.x;
    if (i < n) o[i] = (_Float16)a[i];
}

// ---------------------------------------------------------------------------
// GEMM: C[M,N] = A[M,K](f16) @ W[N,K](f16)^T + bias, optional ReLU.
// Block tile 64x128, 8 waves in 2x4 grid, wave tile 32x32, K-step 32.
// ---------------------------------------------------------------------------
template<int KK, bool RELU, bool OUTF16>
__global__ void k_gemm(const _Float16* __restrict__ A, const _Float16* __restrict__ W,
                       const float* __restrict__ bias, _Float16* __restrict__ Ch,
                       float* __restrict__ Cf, int N) {
    __shared__ alignas(16) _Float16 As[64 * 32];
    __shared__ alignas(16) _Float16 Bs[128 * 32];
    int tid  = threadIdx.x;
    int lane = tid & 31;
    int wave = tid >> 5;
    int wm = wave >> 2;          // 0..1
    int wn = wave & 3;           // 0..3
    int bm = blockIdx.x * 64;
    int bn = blockIdx.y * 128;

    v8f acc[2][2];
    acc[0][0] = (v8f){}; acc[0][1] = (v8f){};
    acc[1][0] = (v8f){}; acc[1][1] = (v8f){};

    int ar = tid >> 2, ac = (tid & 3) * 8;     // A staging: 64 rows x 4 chunks
    int br = tid >> 1, bc = (tid & 1) * 16;    // B staging: 128 rows x 2 chunks

    for (int k0 = 0; k0 < KK; k0 += 32) {
#if USE_ASYNC_LDS
        cp_async16(&As[ar * 32 + ac],     &A[(size_t)(bm + ar) * KK + k0 + ac]);
        cp_async16(&Bs[br * 32 + bc],     &W[(size_t)(bn + br) * KK + k0 + bc]);
        cp_async16(&Bs[br * 32 + bc + 8], &W[(size_t)(bn + br) * KK + k0 + bc + 8]);
        wait_async0();
#else
        *(v8h*)&As[ar * 32 + ac] =
            *(const v8h*)&A[(size_t)(bm + ar) * KK + k0 + ac];
        *(v8h*)&Bs[br * 32 + bc] =
            *(const v8h*)&W[(size_t)(bn + br) * KK + k0 + bc];
        *(v8h*)&Bs[br * 32 + bc + 8] =
            *(const v8h*)&W[(size_t)(bn + br) * KK + k0 + bc + 8];
        if (k0 + 32 < KK) {
            __builtin_prefetch(&A[(size_t)(bm + ar) * KK + k0 + 32 + ac], 0, 1);
            __builtin_prefetch(&W[(size_t)(bn + br) * KK + k0 + 32 + bc], 0, 1);
        }
#endif
        __syncthreads();

        v16h a0 = ld_tile(As + (wm * 32 +  0) * 32, 32, lane);
        v16h a1 = ld_tile(As + (wm * 32 + 16) * 32, 32, lane);
        v16h b0 = ld_tile(Bs + (wn * 32 +  0) * 32, 32, lane);
        v16h b1 = ld_tile(Bs + (wn * 32 + 16) * 32, 32, lane);
        acc[0][0] = wmma_f16(a0, b0, acc[0][0]);
        acc[0][1] = wmma_f16(a0, b1, acc[0][1]);
        acc[1][0] = wmma_f16(a1, b0, acc[1][0]);
        acc[1][1] = wmma_f16(a1, b1, acc[1][1]);
        __syncthreads();
    }

    int lrow = (lane >> 4) * 8;
    int lcol = lane & 15;
    #pragma unroll
    for (int mt = 0; mt < 2; ++mt)
        #pragma unroll
        for (int nt = 0; nt < 2; ++nt)
            #pragma unroll
            for (int r2 = 0; r2 < 8; ++r2) {
                int row = bm + wm * 32 + mt * 16 + lrow + r2;
                int col = bn + wn * 32 + nt * 16 + lcol;
                float v = acc[mt][nt][r2] + bias[col];
                if (RELU) v = fmaxf(v, 0.f);
                if (OUTF16) Ch[(size_t)row * N + col] = (_Float16)v;
                else        Cf[(size_t)row * N + col] = v;
            }
}

// ---------------------------------------------------------------------------
// Flash attention with on-the-fly time bias.
// grid = (B*H, L/64), block = 128 (4 waves, 16 query rows each).
// qkv: [B, L, 3D] f16 (Q at h*32, K at 256+h*32, V at 512+h*32).
// ---------------------------------------------------------------------------
__global__ void k_attn(const _Float16* __restrict__ qkv, const float* __restrict__ t,
                       const unsigned char* __restrict__ mask,
                       const float* __restrict__ time_w, const float* __restrict__ time_b,
                       _Float16* __restrict__ o) {
    __shared__ alignas(16) _Float16 Ks[32 * 32];
    __shared__ alignas(16) _Float16 Vt[32 * 32];   // transposed: Vt[dh][key]
    __shared__ alignas(16) _Float16 Ps[4][16 * 32];

    int b = blockIdx.x / HH;
    int h = blockIdx.x % HH;
    int tid = threadIdx.x, lane = tid & 31, wave = tid >> 5;
    int row0 = blockIdx.y * 64 + wave * 16;

    const float aw = fabsf(time_w[0]);
    const float tb = time_b[0];
    const float scale = 0.17677669529663688f;  // 1/sqrt(32)
    const float NEG = -3.0e38f;

    // Q operand (constant over the key loop)
    v16h qa = ld_tile(qkv + ((size_t)b * LL + row0) * (3 * DD) + h * DHD, 3 * DD, lane);

    int g8 = (lane >> 4) * 8;
    int ncol = lane & 15;
    float trow[8];
    #pragma unroll
    for (int r2 = 0; r2 < 8; ++r2)
        trow[r2] = t[(size_t)b * LL + row0 + g8 + r2];

    float mrow[8], ssum[8];
    v8f o0 = (v8f){}, o1 = (v8f){};
    #pragma unroll
    for (int r2 = 0; r2 < 8; ++r2) { mrow[r2] = NEG; ssum[r2] = 0.f; }

    int sr = tid >> 2, sc = (tid & 3) * 8;     // cooperative staging assignment

    for (int j = 0; j < LL / 32; ++j) {
        int kbase = j * 32;
        // stage K tile (async direct-to-LDS when available) and transposed V tile
        {
            const _Float16* kp =
                qkv + ((size_t)b * LL + kbase + sr) * (3 * DD) + DD + h * DHD + sc;
#if USE_ASYNC_LDS
            cp_async16(&Ks[sr * 32 + sc], kp);
#else
            *(v8h*)&Ks[sr * 32 + sc] = *(const v8h*)kp;
#endif
            const _Float16* vp =
                qkv + ((size_t)b * LL + kbase + sr) * (3 * DD) + 2 * DD + h * DHD + sc;
            v8h vv = *(const v8h*)vp;
            #pragma unroll
            for (int e = 0; e < 8; ++e) Vt[(sc + e) * 32 + sr] = vv[e];
#if USE_ASYNC_LDS
            wait_async0();
#endif
        }
        __syncthreads();

        v16h kb0 = ld_tile(Ks,           32, lane);
        v16h kb1 = ld_tile(Ks + 16 * 32, 32, lane);
        v8f s0 = wmma_f16(qa, kb0, (v8f){});
        v8f s1 = wmma_f16(qa, kb1, (v8f){});

        float tc0 = t[(size_t)b * LL + kbase + ncol];
        float tc1 = t[(size_t)b * LL + kbase + 16 + ncol];
        bool m0 = mask[(size_t)b * LL + kbase + ncol] != 0;
        bool m1 = mask[(size_t)b * LL + kbase + 16 + ncol] != 0;

        float p0[8], p1[8];
        #pragma unroll
        for (int r2 = 0; r2 < 8; ++r2) {
            // bias = tb - |w| * log(1 + |t_i - t_j|): fast v_log_f32 path
            float b0v = tb - aw * __logf(1.0f + fabsf(trow[r2] - tc0));
            float b1v = tb - aw * __logf(1.0f + fabsf(trow[r2] - tc1));
            s0[r2] = m0 ? NEG : s0[r2] * scale + b0v;
            s1[r2] = m1 ? NEG : s1[r2] * scale + b1v;
        }
        #pragma unroll
        for (int r2 = 0; r2 < 8; ++r2) {
            float mx = fmaxf(s0[r2], s1[r2]);
            #pragma unroll
            for (int off = 1; off < 16; off <<= 1)
                mx = fmaxf(mx, __shfl_xor(mx, off, 32));
            float nm = fmaxf(mrow[r2], mx);
            float alpha = __expf(mrow[r2] - nm);
            mrow[r2] = nm;
            float e0 = __expf(s0[r2] - nm);
            float e1 = __expf(s1[r2] - nm);
            float rs = e0 + e1;
            #pragma unroll
            for (int off = 1; off < 16; off <<= 1)
                rs += __shfl_xor(rs, off, 32);
            ssum[r2] = ssum[r2] * alpha + rs;
            o0[r2] *= alpha;
            o1[r2] *= alpha;
            p0[r2] = e0;
            p1[r2] = e1;
        }
        // restage P through wave-private LDS into A-operand layout
        _Float16* ps = &Ps[wave][0];
        #pragma unroll
        for (int r2 = 0; r2 < 8; ++r2) {
            int m = g8 + r2;
            ps[m * 32 + ncol]      = (_Float16)p0[r2];
            ps[m * 32 + 16 + ncol] = (_Float16)p1[r2];
        }
        v16h pa  = ld_tile(ps,           32, lane);
        v16h vb0 = ld_tile(Vt,           32, lane);
        v16h vb1 = ld_tile(Vt + 16 * 32, 32, lane);
        o0 = wmma_f16(pa, vb0, o0);
        o1 = wmma_f16(pa, vb1, o1);
        __syncthreads();
    }

    #pragma unroll
    for (int r2 = 0; r2 < 8; ++r2) {
        float inv = 1.f / ssum[r2];
        size_t orow = ((size_t)b * LL + row0 + g8 + r2) * DD + h * DHD;
        o[orow + ncol]      = (_Float16)(o0[r2] * inv);
        o[orow + 16 + ncol] = (_Float16)(o1[r2] * inv);
    }
}

// ---------------------------------------------------------------------------
// x = LayerNorm(x + r) * g + b   (writes f32 + f16)
// grid = B*L rows, block = 256
// ---------------------------------------------------------------------------
__global__ void k_addln(const float* __restrict__ x, const float* __restrict__ r,
                        const float* __restrict__ g, const float* __restrict__ bta,
                        float* __restrict__ xo, _Float16* __restrict__ xh) {
    __shared__ float red[DD];
    size_t row = blockIdx.x;
    int i = threadIdx.x;
    float v = x[row * DD + i] + r[row * DD + i];
    red[i] = v; __syncthreads();
    for (int off = 128; off > 0; off >>= 1) {
        if (i < off) red[i] += red[i + off];
        __syncthreads();
    }
    float mean = red[0] * (1.f / DD);
    __syncthreads();
    float d = v - mean;
    red[i] = d * d; __syncthreads();
    for (int off = 128; off > 0; off >>= 1) {
        if (i < off) red[i] += red[i + off];
        __syncthreads();
    }
    float var = red[0] * (1.f / DD);
    float y = d * rsqrtf(var + 1e-5f) * g[i] + bta[i];
    xo[row * DD + i] = y;
    xh[row * DD + i] = (_Float16)y;
}

// ---------------------------------------------------------------------------
// Gather last valid state + 2-layer regression head.
// out layout: [0:B) predicted_gap, [B : B + B*D) last_state
// ---------------------------------------------------------------------------
__global__ void k_head(const float* __restrict__ x, const unsigned char* __restrict__ mask,
                       const float* __restrict__ w1, const float* __restrict__ b1,
                       const float* __restrict__ w2, const float* __restrict__ b2,
                       float* __restrict__ out) {
    __shared__ float sred[DD];
    __shared__ float last[DD];
    __shared__ float hbuf[DD / 2];
    int b = blockIdx.x, i = threadIdx.x;
    int c = 0;
    for (int j = i; j < LL; j += DD) c += (mask[(size_t)b * LL + j] == 0) ? 1 : 0;
    sred[i] = (float)c; __syncthreads();
    for (int off = 128; off > 0; off >>= 1) {
        if (i < off) sred[i] += sred[i + off];
        __syncthreads();
    }
    int len = (int)sred[0] - 1;
    float v = x[((size_t)b * LL + len) * DD + i];
    last[i] = v;
    out[BB + (size_t)b * DD + i] = v;
    __syncthreads();
    if (i < DD / 2) {
        float s = b1[i];
        const float* wr = w1 + (size_t)i * DD;
        for (int k2 = 0; k2 < DD; ++k2) s += last[k2] * wr[k2];
        hbuf[i] = fmaxf(s, 0.f);
    }
    __syncthreads();
    if (i == 0) {
        float s = b2[0];
        for (int k2 = 0; k2 < DD / 2; ++k2) s += hbuf[k2] * w2[k2];
        out[b] = s;
    }
}

// ---------------------------------------------------------------------------
// Host launcher
// ---------------------------------------------------------------------------
extern "C" void kernel_launch(void* const* d_in, const int* in_sizes, int n_in,
                              void* d_out, int out_size, void* d_ws, size_t ws_size,
                              hipStream_t stream) {
    (void)in_sizes; (void)n_in; (void)out_size; (void)ws_size;

    const float*         feat   = (const float*)d_in[0];
    const unsigned char* mask   = (const unsigned char*)d_in[1];
    const float*         pe     = (const float*)d_in[2];
    const float*         in_w   = (const float*)d_in[3];
    const float*         in_b   = (const float*)d_in[4];
    const float*         time_w = (const float*)d_in[5];
    const float*         time_b = (const float*)d_in[6];
    const float*         qkv_w  = (const float*)d_in[7];
    const float*         qkv_b  = (const float*)d_in[8];
    const float*         out_w  = (const float*)d_in[9];
    const float*         out_b  = (const float*)d_in[10];
    const float*         ln1_g  = (const float*)d_in[11];
    const float*         ln1_b  = (const float*)d_in[12];
    const float*         ff1_w  = (const float*)d_in[13];
    const float*         ff1_b  = (const float*)d_in[14];
    const float*         ff2_w  = (const float*)d_in[15];
    const float*         ff2_b  = (const float*)d_in[16];
    const float*         ln2_g  = (const float*)d_in[17];
    const float*         ln2_b  = (const float*)d_in[18];
    const float*         reg1_w = (const float*)d_in[19];
    const float*         reg1_b = (const float*)d_in[20];
    const float*         reg2_w = (const float*)d_in[21];
    const float*         reg2_b = (const float*)d_in[22];
    float* out = (float*)d_out;

    // workspace carve-up
    char* base = (char*)d_ws;
    size_t off = 0;
    auto alloc = [&](size_t bytes) -> void* {
        void* p = base + off;
        off += (bytes + 255) & ~(size_t)255;
        return p;
    };
    float*    t     = (float*)   alloc((size_t)BLROWS * 4);
    float*    x     = (float*)   alloc((size_t)BLROWS * DD * 4);
    _Float16* xh    = (_Float16*)alloc((size_t)BLROWS * DD * 2);
    _Float16* qkvh  = (_Float16*)alloc((size_t)BLROWS * 3 * DD * 2);
    _Float16* oh    = (_Float16*)alloc((size_t)BLROWS * DD * 2);
    float*    abuf  = (float*)   alloc((size_t)BLROWS * DD * 4);   // attn/ffn f32 out
    _Float16* ff1h  = (_Float16*)alloc((size_t)BLROWS * DFFD * 2);
    _Float16* wqh   = (_Float16*)alloc((size_t)NLL * 3 * DD * DD * 2);
    _Float16* woh   = (_Float16*)alloc((size_t)NLL * DD * DD * 2);
    _Float16* wf1h  = (_Float16*)alloc((size_t)NLL * DFFD * DD * 2);
    _Float16* wf2h  = (_Float16*)alloc((size_t)NLL * DD * DFFD * 2);

    // weight conversion f32 -> f16
    int nq  = NLL * 3 * DD * DD;
    int no2 = NLL * DD * DD;
    int nf1 = NLL * DFFD * DD;
    int nf2 = NLL * DD * DFFD;
    k_f32toh<<<(nq  + 255) / 256, 256, 0, stream>>>(qkv_w, wqh,  nq);
    k_f32toh<<<(no2 + 255) / 256, 256, 0, stream>>>(out_w, woh,  no2);
    k_f32toh<<<(nf1 + 255) / 256, 256, 0, stream>>>(ff1_w, wf1h, nf1);
    k_f32toh<<<(nf2 + 255) / 256, 256, 0, stream>>>(ff2_w, wf2h, nf2);

    k_cumsum<<<BB, LL, 0, stream>>>(feat, t);
    k_input<<<(BLROWS * DD) / 256, 256, 0, stream>>>(feat, in_w, in_b, pe, x, xh);

    for (int l = 0; l < NLL; ++l) {
        // QKV projection: [16384,256] x [768,256]^T -> f16
        k_gemm<DD, false, true><<<dim3(BLROWS / 64, (3 * DD) / 128), 256, 0, stream>>>(
            xh, wqh + (size_t)l * 3 * DD * DD, qkv_b + (size_t)l * 3 * DD,
            qkvh, nullptr, 3 * DD);

        // flash attention -> oh (f16)
        k_attn<<<dim3(BB * HH, LL / 64), 128, 0, stream>>>(
            qkvh, t, mask, time_w, time_b, oh);

        // output projection -> abuf (f32)
        k_gemm<DD, false, false><<<dim3(BLROWS / 64, DD / 128), 256, 0, stream>>>(
            oh, woh + (size_t)l * DD * DD, out_b + (size_t)l * DD,
            nullptr, abuf, DD);

        // x = LN(x + attn)
        k_addln<<<BLROWS, DD, 0, stream>>>(
            x, abuf, ln1_g + (size_t)l * DD, ln1_b + (size_t)l * DD, x, xh);

        // FF1 with ReLU -> f16
        k_gemm<DD, true, true><<<dim3(BLROWS / 64, DFFD / 128), 256, 0, stream>>>(
            xh, wf1h + (size_t)l * DFFD * DD, ff1_b + (size_t)l * DFFD,
            ff1h, nullptr, DFFD);

        // FF2 -> abuf (f32)
        k_gemm<DFFD, false, false><<<dim3(BLROWS / 64, DD / 128), 256, 0, stream>>>(
            ff1h, wf2h + (size_t)l * DD * DFFD, ff2_b + (size_t)l * DD,
            nullptr, abuf, DD);

        // x = LN(x + ffn)
        k_addln<<<BLROWS, DD, 0, stream>>>(
            x, abuf, ln2_g + (size_t)l * DD, ln2_b + (size_t)l * DD, x, xh);
    }

    k_head<<<BB, DD, 0, stream>>>(x, mask, reg1_w, reg1_b, reg2_w, reg2_b, out);
}
